// AlignSubLayer_21818433864125
// MI455X (gfx1250) — matrix operand order repaired and checked by reading, not compiled, not run on hardware
//
#include <hip/hip_runtime.h>

typedef __attribute__((ext_vector_type(16))) _Float16 v16h;
typedef __attribute__((ext_vector_type(8)))  _Float16 v8h;
typedef __attribute__((ext_vector_type(4)))  _Float16 v4h;
typedef __attribute__((ext_vector_type(8)))  float    v8f;

#define WMMA_F16(a, b, c) \
  __builtin_amdgcn_wmma_f32_16x16x32_f16(false, (a), false, (b), (short)0, (c), false, false)

constexpr int BATCH = 128;
constexpr int L     = 512;   // LC == LM
constexpr int D     = 512;
constexpr int NT    = 256;   // threads per block (8 wave32)

// Shared memory, unioned across the two phases (~127 KB total)
union Shared {
  struct {
    _Float16 mainT[64 * 512];   // 64KB  : f16 chunk of main rows
    _Float16 ctx  [32 * 512];   // 32KB  : f16 chunk of context rows
    float    sc   [32 * 64];    //  8KB  : score tile staging
    float    red  [32 * 9];     //  ~1KB : row reduction scratch
  } p1;
  struct {
    _Float16 mainT  [16 * 512]; // 16KB  : f16 m-tile of main
    _Float16 ctx    [32 * 512]; // 32KB  : f16 context chunk (row major)
    _Float16 ctxT   [512 * 32]; // 32KB  : same chunk, transposed [d][c]
    _Float16 attnT  [16 * 32];  //  1KB  : attn tile, m-major
    float    score  [32 * 16];  //  2KB  : cooperative score partials
    float    aligned[16 * 512]; // 32KB  : aligned tile -> diff tile
    float    red    [16 * 16];  //  1KB  : logit partials
    float    p      [16];       // softmax weights of this m-tile
    float    logit  [16];
  } p2;
};

// A fragment: element [row0 + (lane&15)][k]; halves 0..15 at k0+16*hi..+15.
// Both runs are 16B-aligned contiguous -> 2x ds_load_b128.
__device__ __forceinline__ v16h load_fragA(const _Float16* base, int ld, int row0, int k0, int lane) {
  const _Float16* p = base + (row0 + (lane & 15)) * ld + k0 + 16 * (lane >> 4);
  const v8h lo = *reinterpret_cast<const v8h*>(p);
  const v8h hi = *reinterpret_cast<const v8h*>(p + 8);
  return __builtin_shufflevector(lo, hi, 0, 1, 2, 3, 4, 5, 6, 7, 8, 9, 10, 11, 12, 13, 14, 15);
}

// B fragment: element [k][n0 + (lane&15)], stored k-contiguous per n-row; halves k = k0+16*hi..+15.
// One 32B-aligned contiguous run -> 2x ds_load_b128.
__device__ __forceinline__ v16h load_fragB(const _Float16* base, int ld, int n0, int k0, int lane) {
  const _Float16* p = base + (n0 + (lane & 15)) * ld + k0 + 16 * (lane >> 4);
  return *reinterpret_cast<const v16h*>(p);
}

__device__ __forceinline__ void store4h(_Float16* dst, const float4 v) {
  *reinterpret_cast<v4h*>(dst) =
      (v4h){(_Float16)v.x, (_Float16)v.y, (_Float16)v.z, (_Float16)v.w};
}

__global__ __launch_bounds__(NT, 1) void align_sub_layer_kernel(
    const float* __restrict__ g_ctx, const float* __restrict__ g_main,
    const float* __restrict__ g_W, float* __restrict__ g_out)
{
  __shared__ Shared u;
  __shared__ float s_rowmax[L];
  __shared__ float s_rowsum[L];
  __shared__ float s_W[D];
  __shared__ float s_out[D];
  __shared__ float s_run[3];   // 0: running max, 1: running sum, 2: rescale factor

  const int tid  = threadIdx.x;
  const int lane = tid & 31;
  const int w    = tid >> 5;
  const int b    = blockIdx.x;
  const float* ctx = g_ctx  + (size_t)b * L * D;
  const float* mn  = g_main + (size_t)b * L * D;

  for (int i = tid; i < L; i += NT) { s_rowmax[i] = -1e30f; s_rowsum[i] = 0.f; }
  for (int i = tid; i < D; i += NT) { s_W[i] = g_W[i]; s_out[i] = 0.f; }
  if (tid == 0) { s_run[0] = -1e30f; s_run[1] = 0.f; s_run[2] = 1.f; }
  __syncthreads();

  const v8f Z = {0.f, 0.f, 0.f, 0.f, 0.f, 0.f, 0.f, 0.f};

  // ======================= Phase 1: per-row softmax statistics over m ====================
  for (int mc = 0; mc < L / 64; ++mc) {
    // stage 64 rows of main as f16
    for (int i = tid * 4; i < 64 * D; i += NT * 4) {
      const float4 v = *reinterpret_cast<const float4*>(mn + (size_t)mc * 64 * D + i);
      store4h(u.p1.mainT + i, v);
    }
    __syncthreads();
    for (int cp = 0; cp < L / 32; ++cp) {
      const int c0 = cp * 32;
      // stage 32 rows of context as f16
      for (int i = tid * 4; i < 32 * D; i += NT * 4) {
        const float4 v = *reinterpret_cast<const float4*>(ctx + (size_t)c0 * D + i);
        store4h(u.p1.ctx + i, v);
      }
      __syncthreads();

      // each wave: one 16x16 score tile, K = D = 512 (16 k-steps)
      const int csub = w >> 2, nt = w & 3;
      v8f acc = Z;
#pragma unroll
      for (int ks = 0; ks < 16; ++ks) {
        v16h a  = load_fragA(u.p1.ctx,   D, csub * 16, ks * 32, lane);
        v16h bm = load_fragB(u.p1.mainT, D, nt * 16,   ks * 32, lane);
        acc = WMMA_F16(a, bm, acc);
      }
#pragma unroll
      for (int i = 0; i < 8; ++i)
        u.p1.sc[(csub * 16 + i + 8 * (lane >> 4)) * 64 + nt * 16 + (lane & 15)] = acc[i];
      __syncthreads();

      // online softmax stats for rows c0..c0+31 over these 64 m columns
      {
        const int r = tid >> 3, j = tid & 7;
        float pm = -1e30f;
        for (int q = 0; q < 8; ++q) pm = fmaxf(pm, u.p1.sc[r * 64 + j * 8 + q]);
        u.p1.red[r * 9 + j] = pm;
        __syncthreads();
        if (j == 0) {
          float tm = u.p1.red[r * 9];
          for (int q = 1; q < 8; ++q) tm = fmaxf(tm, u.p1.red[r * 9 + q]);
          u.p1.red[r * 9 + 8] = fmaxf(tm, s_rowmax[c0 + r]);
        }
        __syncthreads();
        const float nmax = u.p1.red[r * 9 + 8];
        float ps = 0.f;
        for (int q = 0; q < 8; ++q) ps += __expf(u.p1.sc[r * 64 + j * 8 + q] - nmax);
        u.p1.red[r * 9 + j] = ps;
        __syncthreads();
        if (j == 0) {
          float ts = 0.f;
          for (int q = 0; q < 8; ++q) ts += u.p1.red[r * 9 + q];
          const float om = s_rowmax[c0 + r];
          s_rowsum[c0 + r] = s_rowsum[c0 + r] * __expf(om - nmax) + ts;
          s_rowmax[c0 + r] = nmax;
        }
        __syncthreads();
      }
    }
    __syncthreads();
  }

  // ============== Phase 2: aligned, diff, projection, online output softmax ==============
  for (int mt = 0; mt < L / 16; ++mt) {
    const int m0 = mt * 16;
    // stage m-tile of main as f16
    for (int i = tid * 4; i < 16 * D; i += NT * 4) {
      const float4 v = *reinterpret_cast<const float4*>(mn + (size_t)m0 * D + i);
      store4h(u.p2.mainT + i, v);
    }
    v8f al[4] = {Z, Z, Z, Z};    // aligned accumulators: wave w owns d columns [w*64, w*64+64)

    for (int cc = 0; cc < L / 32; ++cc) {
      const int c0 = cc * 32;
      __syncthreads();
      // stage context chunk, row-major + transposed
      for (int i = tid * 4; i < 32 * D; i += NT * 4) {
        const float4 v = *reinterpret_cast<const float4*>(ctx + (size_t)c0 * D + i);
        const int c = i >> 9;
        const int d = i & 511;
        const _Float16 h0 = (_Float16)v.x, h1 = (_Float16)v.y;
        const _Float16 h2 = (_Float16)v.z, h3 = (_Float16)v.w;
        *reinterpret_cast<v4h*>(u.p2.ctx + i) = (v4h){h0, h1, h2, h3};
        u.p2.ctxT[(d + 0) * 32 + c] = h0;
        u.p2.ctxT[(d + 1) * 32 + c] = h1;
        u.p2.ctxT[(d + 2) * 32 + c] = h2;
        u.p2.ctxT[(d + 3) * 32 + c] = h3;
      }
      for (int i = tid; i < 32 * 16; i += NT) u.p2.score[i] = 0.f;
      __syncthreads();

      // recompute scores[c0..c0+31, m-tile]; K split across waves (64 each), partials via ds_add
      v8f s0 = Z, s1 = Z;
#pragma unroll
      for (int ks = 0; ks < 2; ++ks) {
        const int k0 = w * 64 + ks * 32;
        v16h bm = load_fragB(u.p2.mainT, D, 0, k0, lane);
        v16h a0 = load_fragA(u.p2.ctx, D, 0, k0, lane);
        s0 = WMMA_F16(a0, bm, s0);
        v16h a1 = load_fragA(u.p2.ctx, D, 16, k0, lane);
        s1 = WMMA_F16(a1, bm, s1);
      }
#pragma unroll
      for (int i = 0; i < 8; ++i) {
        atomicAdd(&u.p2.score[(i + 8 * (lane >> 4)) * 16 + (lane & 15)], s0[i]);
        atomicAdd(&u.p2.score[(16 + i + 8 * (lane >> 4)) * 16 + (lane & 15)], s1[i]);
      }
      __syncthreads();

      // normalize -> f16 attn^T tile [m][c]
      for (int i = tid; i < 32 * 16; i += NT) {
        const int c = i >> 4, m = i & 15;
        const float a = __expf(u.p2.score[i] - s_rowmax[c0 + c]) / s_rowsum[c0 + c];
        u.p2.attnT[m * 32 + c] = (_Float16)a;
      }
      __syncthreads();

      // aligned[m-tile, :] += attn^T (16x32) x ctx (32xD)
      v16h af = load_fragA(u.p2.attnT, 32, 0, 0, lane);
#pragma unroll
      for (int t = 0; t < 4; ++t) {
        v16h bf = load_fragB(u.p2.ctxT, 32, (w * 4 + t) * 16, 0, lane);
        al[t] = WMMA_F16(af, bf, al[t]);
      }
    }
    __syncthreads();

    // spill aligned accumulators to LDS
#pragma unroll
    for (int t = 0; t < 4; ++t)
#pragma unroll
      for (int i = 0; i < 8; ++i)
        u.p2.aligned[(i + 8 * (lane >> 4)) * D + (w * 4 + t) * 16 + (lane & 15)] = al[t][i];
    __syncthreads();

    // diff = main - aligned (in place), partial logits = diff . W
    {
      const int mrow = tid >> 4, jj = tid & 15;
      float plg = 0.f;
      for (int q = 0; q < 32; ++q) {
        const int d = jj * 32 + q;
        const float df = mn[(size_t)(m0 + mrow) * D + d] - u.p2.aligned[mrow * D + d];
        u.p2.aligned[mrow * D + d] = df;
        plg += df * s_W[d];
      }
      u.p2.red[mrow * 16 + jj] = plg;
    }
    __syncthreads();
    if (tid < 16) {
      float lg = 0.f;
      for (int q = 0; q < 16; ++q) lg += u.p2.red[tid * 16 + q];
      u.p2.logit[tid] = lg;
    }
    __syncthreads();
    if (tid == 0) {
      float tm = u.p2.logit[0];
      for (int q = 1; q < 16; ++q) tm = fmaxf(tm, u.p2.logit[q]);
      const float om = s_run[0];
      const float nm = fmaxf(om, tm);
      const float scale = __expf(om - nm);
      float psum = 0.f;
      for (int q = 0; q < 16; ++q) {
        const float e = __expf(u.p2.logit[q] - nm);
        u.p2.p[q] = e;
        psum += e;
      }
      s_run[1] = s_run[1] * scale + psum;
      s_run[0] = nm;
      s_run[2] = scale;
    }
    __syncthreads();
    {
      const float scale = s_run[2];
      for (int d = tid; d < D; d += NT) {
        float o = s_out[d] * scale;
        for (int q = 0; q < 16; ++q) o += u.p2.p[q] * u.p2.aligned[q * D + d];
        s_out[d] = o;
      }
    }
    __syncthreads();
  }

  // epilogue: normalize by running sum and store out[b, :]
  {
    const float inv = 1.0f / s_run[1];
    for (int d = tid; d < D; d += NT) g_out[(size_t)b * D + d] = s_out[d] * inv;
  }
}

extern "C" void kernel_launch(void* const* d_in, const int* in_sizes, int n_in,
                              void* d_out, int out_size, void* d_ws, size_t ws_size,
                              hipStream_t stream) {
  (void)in_sizes; (void)n_in; (void)out_size; (void)d_ws; (void)ws_size;
  const float* ctx = (const float*)d_in[0];
  const float* mn  = (const float*)d_in[1];
  const float* W   = (const float*)d_in[2];
  float* out = (float*)d_out;
  align_sub_layer_kernel<<<dim3(BATCH), dim3(NT), 0, stream>>>(ctx, mn, W, out);
}
